// MultiheadAttention_26010321944681
// MI455X (gfx1250) — matrix operand order repaired and checked by reading, not compiled, not run on hardware
//
#include <hip/hip_runtime.h>

typedef __bf16 bf16_t;
typedef __attribute__((ext_vector_type(8)))  __bf16 v8bf;
typedef __attribute__((ext_vector_type(16))) __bf16 v16bf;
typedef __attribute__((ext_vector_type(8)))  float  v8f;
typedef __attribute__((ext_vector_type(4)))  unsigned uint4v;
typedef __attribute__((ext_vector_type(8)))  int      int8v;
typedef __attribute__((ext_vector_type(4)))  int      int4v;

#define DIMX   1024
#define NHEAD  16
#define HDIM   64
#define BATCH  2
#define SEQ    2048
#define MTOT   (BATCH * SEQ)   // 4096

// ---------- helpers ----------

__device__ __forceinline__ bf16_t f2bf(float f) {
  union { float f; unsigned u; } a; a.f = f;
  unsigned r = a.u + 0x7fffu + ((a.u >> 16) & 1u);   // round-to-nearest-even
  union { unsigned short s; bf16_t b; } o; o.s = (unsigned short)(r >> 16);
  return o.b;
}

__device__ __forceinline__ v8f zero8() {
  v8f z = {0.f, 0.f, 0.f, 0.f, 0.f, 0.f, 0.f, 0.f};
  return z;
}

__device__ __forceinline__ v16bf load_frag(const bf16_t* p0, const bf16_t* p1) {
  v8bf lo = *(const v8bf*)p0;
  v8bf hi = *(const v8bf*)p1;
  return __builtin_shufflevector(lo, hi, 0, 1, 2, 3, 4, 5, 6, 7,
                                          8, 9, 10, 11, 12, 13, 14, 15);
}

__device__ __forceinline__ v8f wmma_bf16(v16bf a, v16bf b, v8f c) {
  return __builtin_amdgcn_wmma_f32_16x16x32_bf16(
      false, a, false, b, (short)0, c, false, false);
}

// TDM: DMA a 2-D bf16 tile (tile_d0 contiguous elems x tile_d1 rows) from global
// memory (row pitch = stride elems) into LDS at byte offset lds_off.
// Descriptor bit packing per CDNA5 ISA 8.3/8.4 (count=1, type=2, data_size=2B).
__device__ __forceinline__ void tdm_load_2d(unsigned lds_off, const bf16_t* gptr,
                                            unsigned tensor_d0, unsigned tensor_d1,
                                            unsigned stride_d0,
                                            unsigned tile_d0, unsigned tile_d1) {
  unsigned long long ga = (unsigned long long)(uintptr_t)gptr;
  uint4v g0;
  g0[0] = 1u;                                          // count=1, no gather
  g0[1] = lds_off;                                     // lds_addr
  g0[2] = (unsigned)(ga & 0xffffffffu);                // global_addr[31:0]
  g0[3] = (unsigned)((ga >> 32) & 0x01ffffffu) | 0x80000000u; // addr[56:32] | type=2
  int8v g1;
  g1[0] = 0x10000;                                     // data_size=1 (2 bytes)
  g1[1] = (int)((tensor_d0 & 0xffffu) << 16);          // tensor_dim0[15:0] @ bit48
  g1[2] = (int)((tensor_d0 >> 16) | ((tensor_d1 & 0xffffu) << 16));
  g1[3] = (int)((tensor_d1 >> 16) | (tile_d0 << 16));  // tile_dim0 @ bit112
  g1[4] = (int)tile_d1;                                // tile_dim1 (tile_dim2=0)
  g1[5] = (int)stride_d0;                              // tensor_dim0_stride[31:0]
  g1[6] = 0;
  g1[7] = 0;
  int4v gz4 = {0, 0, 0, 0};
  int8v gz8 = {0, 0, 0, 0, 0, 0, 0, 0};
  __builtin_amdgcn_tensor_load_to_lds(g0, g1, gz4, gz4, gz8, 0);
}

// ---------- precision conversion kernels ----------

__global__ void k_cvt(const float* __restrict__ in, bf16_t* __restrict__ out, int n) {
  int i = blockIdx.x * blockDim.x + threadIdx.x;
  if (i < n) out[i] = f2bf(in[i]);
}

// in: [K, N] row-major fp32 -> out: [N, K] row-major bf16
__global__ void k_transpose_cvt(const float* __restrict__ in, bf16_t* __restrict__ out,
                                int K, int N) {
  int i = blockIdx.x * blockDim.x + threadIdx.x;
  if (i < N * K) {
    int n = i / K;
    int k = i - n * K;
    out[i] = f2bf(in[k * N + n]);
  }
}

// ---------- shared GEMM core: 128x64 block tile, TDM double-buffered LDS ----------
// LDS layout (dynamic): A0 @0 (8KB), A1 @8192, B0 @16384 (4KB), B1 @20480 -> 24KB

#define GEMM_LDS_BYTES 24576

// computes acc[4] (wave's 16x64 C tile) for C = A[M,1024] @ Bt[N,1024]^T
__device__ __forceinline__ void gemm_core(const bf16_t* __restrict__ A,
                                          const bf16_t* __restrict__ Bt,
                                          int m_base, int n_base, int n_rows_total,
                                          int wave, int half, int l16, v8f acc[4]) {
  extern __shared__ __align__(16) char smem[];
  bf16_t* ldsA = (bf16_t*)smem;              // [2][128][32]
  bf16_t* ldsB = (bf16_t*)(smem + 16384);    // [2][64][32]

  for (int t = 0; t < 4; ++t) acc[t] = zero8();

  if (wave == 0) {  // prologue: fill buffer 0
    tdm_load_2d(0,     A  + (size_t)m_base * DIMX, DIMX, MTOT,         DIMX, 32, 128);
    tdm_load_2d(16384, Bt + (size_t)n_base * DIMX, DIMX, (unsigned)n_rows_total,
                DIMX, 32, 64);
  }

  for (int it = 0; it < DIMX / 32; ++it) {
    const int buf = it & 1;
    const int k0n = (it + 1) * 32;
    if (wave == 0) {
      if (k0n < DIMX) {   // issue next tile pair into the other buffer
        tdm_load_2d((unsigned)((buf ^ 1) * 8192),
                    A + (size_t)m_base * DIMX + k0n, DIMX, MTOT, DIMX, 32, 128);
        tdm_load_2d((unsigned)(16384 + (buf ^ 1) * 4096),
                    Bt + (size_t)n_base * DIMX + k0n, DIMX, (unsigned)n_rows_total,
                    DIMX, 32, 64);
        __builtin_amdgcn_s_wait_tensorcnt(2);   // current buffer's pair done
      } else {
        __builtin_amdgcn_s_wait_tensorcnt(0);
      }
    }
    __syncthreads();

    const bf16_t* a_tile = ldsA + buf * 4096;  // 128*32
    const bf16_t* b_tile = ldsB + buf * 2048;  // 64*32
    const int am  = wave * 16 + l16;
    const int akb = half * 8;
    v16bf af = load_frag(a_tile + am * 32 + akb, a_tile + am * 32 + akb + 16);
    for (int t = 0; t < 4; ++t) {
      const bf16_t* bp = b_tile + (t * 16 + l16) * 32 + half * 8;
      v16bf bfr = load_frag(bp, bp + 16);
      acc[t] = wmma_bf16(af, bfr, acc[t]);
    }
    __syncthreads();   // all reads done before this buffer is refilled
  }
}

// ---------- fused QKV GEMM: epilogue scatters into Q [bh,S,HD], K [bh,S,HD], Vt [bh,HD,S]

__global__ __launch_bounds__(256) void k_qkv_gemm(
    const bf16_t* __restrict__ A, const bf16_t* __restrict__ Bt,
    const float* __restrict__ bias,
    bf16_t* __restrict__ Qb, bf16_t* __restrict__ Kb, bf16_t* __restrict__ Vtb) {
  const int tid  = threadIdx.x;
  const int wave = tid >> 5;
  const int lane = tid & 31;
  const int half = lane >> 4;
  const int l16  = lane & 15;
  const int m_base = blockIdx.y * 128;
  const int n_base = blockIdx.x * 64;

  v8f acc[4];
  gemm_core(A, Bt, m_base, n_base, 3 * DIMX, wave, half, l16, acc);

  const int row0 = m_base + wave * 16 + half * 8;
  for (int t = 0; t < 4; ++t) {
    const int ncol = n_base + t * 16 + l16;
    const float bv = bias[ncol];
    const int sel  = ncol >> 10;         // 0=q 1=k 2=v
    const int rem  = ncol & 1023;
    const int head = rem >> 6;
    const int hd   = rem & 63;
    for (int r = 0; r < 8; ++r) {
      const int rg = row0 + r;
      const int b  = rg >> 11;           // / SEQ
      const int s  = rg & 2047;
      const bf16_t v = f2bf(acc[t][r] + bv);
      const size_t bh = (size_t)b * NHEAD + head;
      if (sel == 0)      Qb [(bh * SEQ + s) * HDIM + hd] = v;
      else if (sel == 1) Kb [(bh * SEQ + s) * HDIM + hd] = v;
      else               Vtb[(bh * HDIM + hd) * SEQ + s] = v;
    }
  }
}

// ---------- output projection: Out[4096,1024] fp32 ----------

__global__ __launch_bounds__(256) void k_proj_gemm(
    const bf16_t* __restrict__ A, const bf16_t* __restrict__ Bt,
    const float* __restrict__ bias, float* __restrict__ Out) {
  const int tid  = threadIdx.x;
  const int wave = tid >> 5;
  const int lane = tid & 31;
  const int half = lane >> 4;
  const int l16  = lane & 15;
  const int m_base = blockIdx.y * 128;
  const int n_base = blockIdx.x * 64;

  v8f acc[4];
  gemm_core(A, Bt, m_base, n_base, DIMX, wave, half, l16, acc);

  const int row0 = m_base + wave * 16 + half * 8;
  for (int t = 0; t < 4; ++t) {
    const int ncol = n_base + t * 16 + l16;
    const float bv = bias[ncol];
    for (int r = 0; r < 8; ++r) {
      const int rg = row0 + r;
      Out[(size_t)rg * DIMX + ncol] = acc[t][r] + bv;
    }
  }
}

// ---------- flash-style attention: one workgroup = 64 query rows of one (b,h)

__global__ __launch_bounds__(128) void k_attn(
    const bf16_t* __restrict__ Qb, const bf16_t* __restrict__ Kb,
    const bf16_t* __restrict__ Vtb, bf16_t* __restrict__ Ob /* [B,S,DIM] bf16 */) {
  __shared__ __align__(16) bf16_t lds_p[4][16][64];

  const int tid  = threadIdx.x;
  const int wave = tid >> 5;
  const int lane = tid & 31;
  const int half = lane >> 4;
  const int l16  = lane & 15;

  const int bh = blockIdx.y;            // 0..31
  const int b  = bh >> 4;
  const int h  = bh & 15;
  const int q0 = blockIdx.x * 64;

  const bf16_t* Qh = Qb  + (size_t)bh * SEQ * HDIM;
  const bf16_t* Kh = Kb  + (size_t)bh * SEQ * HDIM;
  const bf16_t* Vh = Vtb + (size_t)bh * HDIM * SEQ;

  // preload this wave's 16x64 Q tile as two A fragments (K-steps 0 and 32)
  const int mrow = q0 + wave * 16 + l16;
  v16bf qa0, qa1;
  {
    const bf16_t* qp = Qh + (size_t)mrow * HDIM;
    const int kb = half * 8;
    qa0 = load_frag(qp + kb,      qp + kb + 16);
    qa1 = load_frag(qp + 32 + kb, qp + 32 + kb + 16);
  }

  v8f acc[4];
  for (int t = 0; t < 4; ++t) acc[t] = zero8();
  float mrun[8], lrun[8];
  for (int r = 0; r < 8; ++r) { mrun[r] = -3.0e38f; lrun[r] = 0.f; }

  for (int j0 = 0; j0 < SEQ; j0 += 64) {
    // ---- scores tile 16x64 = Q (16x64) @ K^T ----
    v8f st[4];
    for (int t = 0; t < 4; ++t) {
      const bf16_t* kp = Kh + (size_t)(j0 + t * 16 + l16) * HDIM + half * 8;
      __builtin_prefetch(kp + 64 * HDIM, 0, 0);     // next key block
      v16bf kf0 = load_frag(kp,      kp + 16);      // head-dim 0..31
      v16bf kf1 = load_frag(kp + 32, kp + 48);      // head-dim 32..63
      st[t] = wmma_bf16(qa0, kf0, zero8());
      st[t] = wmma_bf16(qa1, kf1, st[t]);
    }

    // ---- online softmax (scale = 1/sqrt(64) = 0.125) ----
    float mnew[8], alpha[8];
    for (int r = 0; r < 8; ++r) {
      float v = fmaxf(fmaxf(st[0][r], st[1][r]), fmaxf(st[2][r], st[3][r]));
      v = fmaxf(v, __shfl_xor(v, 1, 32));
      v = fmaxf(v, __shfl_xor(v, 2, 32));
      v = fmaxf(v, __shfl_xor(v, 4, 32));
      v = fmaxf(v, __shfl_xor(v, 8, 32));
      v *= 0.125f;
      mnew[r]  = fmaxf(mrun[r], v);
      alpha[r] = __expf(mrun[r] - mnew[r]);
      mrun[r]  = mnew[r];
    }
    for (int t = 0; t < 4; ++t)
      for (int r = 0; r < 8; ++r) {
        float p = __expf(st[t][r] * 0.125f - mnew[r]);
        st[t][r] = p;
        lds_p[wave][half * 8 + r][t * 16 + l16] = f2bf(p);  // C-layout -> row-major
      }
    for (int r = 0; r < 8; ++r) {
      float s = st[0][r] + st[1][r] + st[2][r] + st[3][r];
      s += __shfl_xor(s, 1, 32);
      s += __shfl_xor(s, 2, 32);
      s += __shfl_xor(s, 4, 32);
      s += __shfl_xor(s, 8, 32);
      lrun[r] = lrun[r] * alpha[r] + s;
      for (int t = 0; t < 4; ++t) acc[t][r] *= alpha[r];
    }
    __syncthreads();

    // ---- acc += P (16x64) @ V (64x64) ----
    for (int ks = 0; ks < 2; ++ks) {
      const int kb = half * 8;
      const bf16_t* pp = &lds_p[wave][l16][ks * 32 + kb];
      v16bf pf = load_frag(pp, pp + 16);
      for (int t = 0; t < 4; ++t) {
        const bf16_t* vp = Vh + (size_t)(t * 16 + l16) * SEQ + j0 + ks * 32 + half * 8;
        __builtin_prefetch(vp + 64, 0, 0);          // next key block
        v16bf vf = load_frag(vp, vp + 16);
        acc[t] = wmma_bf16(pf, vf, acc[t]);
      }
    }
    __syncthreads();   // protect lds_p before next iteration's writes
  }

  // ---- epilogue: normalize and scatter into [B,S,DIM] ----
  const int srow0 = q0 + wave * 16 + half * 8;
  for (int r = 0; r < 8; ++r) {
    const float inv = 1.0f / lrun[r];
    const int s = srow0 + r;
    bf16_t* op = Ob + ((size_t)b * SEQ + s) * DIMX + h * HDIM;
    for (int t = 0; t < 4; ++t)
      op[t * 16 + l16] = f2bf(acc[t][r] * inv);
  }
}

// ---------- host launch ----------

extern "C" void kernel_launch(void* const* d_in, const int* in_sizes, int n_in,
                              void* d_out, int out_size, void* d_ws, size_t ws_size,
                              hipStream_t stream) {
  const float* x      = (const float*)d_in[0];
  const float* w_qkv  = (const float*)d_in[1];
  const float* b_qkv  = (const float*)d_in[2];
  const float* w_proj = (const float*)d_in[3];
  const float* b_proj = (const float*)d_in[4];
  float* out = (float*)d_out;

  char* ws = (char*)d_ws;
  size_t off = 0;
  bf16_t* xb     = (bf16_t*)(ws + off); off += (size_t)MTOT * DIMX * 2;        // 8 MB
  bf16_t* wqkvT  = (bf16_t*)(ws + off); off += (size_t)3 * DIMX * DIMX * 2;    // 6 MB
  bf16_t* wprojT = (bf16_t*)(ws + off); off += (size_t)DIMX * DIMX * 2;        // 2 MB
  bf16_t* Qb     = (bf16_t*)(ws + off); off += (size_t)MTOT * DIMX * 2;        // 8 MB
  bf16_t* Kb     = (bf16_t*)(ws + off); off += (size_t)MTOT * DIMX * 2;        // 8 MB
  bf16_t* Vtb    = (bf16_t*)(ws + off); off += (size_t)MTOT * DIMX * 2;        // 8 MB
  bf16_t* attnb  = (bf16_t*)(ws + off); off += (size_t)MTOT * DIMX * 2;        // 8 MB

  const int n_x = MTOT * DIMX;
  k_cvt<<<(n_x + 255) / 256, 256, 0, stream>>>(x, xb, n_x);
  const int n_wq = 3 * DIMX * DIMX;
  k_transpose_cvt<<<(n_wq + 255) / 256, 256, 0, stream>>>(w_qkv, wqkvT, DIMX, 3 * DIMX);
  const int n_wp = DIMX * DIMX;
  k_transpose_cvt<<<(n_wp + 255) / 256, 256, 0, stream>>>(w_proj, wprojT, DIMX, DIMX);

  k_qkv_gemm<<<dim3(3 * DIMX / 64, MTOT / 128), 256, GEMM_LDS_BYTES, stream>>>(
      xb, wqkvT, b_qkv, Qb, Kb, Vtb);

  k_attn<<<dim3(SEQ / 64, BATCH * NHEAD), 128, 0, stream>>>(Qb, Kb, Vtb, attnb);

  k_proj_gemm<<<dim3(DIMX / 64, MTOT / 128), 256, GEMM_LDS_BYTES, stream>>>(
      attnb, wprojT, b_proj, out);
}
